// DynamicModel_49331994362163
// MI455X (gfx1250) — matrix operand order repaired and checked by reading, not compiled, not run on hardware
//
#include <hip/hip_runtime.h>

// Problem constants (match reference)
#define B_SZ   128
#define NIN    4
#define NOUT   4
#define T_SZ   2048
#define RF     64
#define HID    256
#define NCH    (NIN + NOUT)   // 8
#define KDIM   256            // 4 channels * RF  (per-half K)
#define KT_TILES 8            // KDIM/32
#define NT_TILES 16           // HID/16
#define FRAG_DWORDS (KT_TILES * NT_TILES * 32 * 8)   // 32768 dwords = 128KB

typedef __attribute__((ext_vector_type(16))) _Float16 v16h;
typedef __attribute__((ext_vector_type(8)))  float    v8f;

union FragU { unsigned u[8]; uint4 q[2]; v16h v; };
union AccU  { float    f[8]; v8f  v; };

__device__ __forceinline__ unsigned f2h_bits(float x) {
  _Float16 h = (_Float16)x;
  union { _Float16 h; unsigned short s; } cv; cv.h = h;
  return (unsigned)cv.s;
}

// ---- CDNA5 async global->LDS copy (ASYNCcnt-tracked, ISA 15.18.3 op 98) ----
__device__ __forceinline__ void async_g2l_b128(unsigned lds_off, unsigned gl_off,
                                               const void* sbase) {
  asm volatile("global_load_async_to_lds_b128 %0, %1, %2"
               :: "v"(lds_off), "v"(gl_off), "s"(sbase)
               : "memory");
}
__device__ __forceinline__ void wait_async0() {
  asm volatile("s_wait_asynccnt 0" ::: "memory");
}

// A fragment (16-bit 16x32) from the LDS y-history ring.
// Lanes 0-15 hold Klocal 0-7 (v0..3) and 16-23 (v4..7); lanes 16-31: 8-15/24-31.
__device__ __forceinline__ void load_a_ring(const unsigned short* __restrict__ ybuf,
                                            int brow, int base, int hi, int kt,
                                            FragU& dst) {
#pragma unroll
  for (int v = 0; v < 8; ++v) {
    int klocal = (hi ? 8 : 0) + ((v >= 4) ? 16 : 0) + 2 * (v & 3);
    int kg = kt * 32 + klocal;
    int c  = kg >> 6;
    int kk = kg & 63;
    const unsigned short* yp = &ybuf[((c * B_SZ + brow) * 128) + base + kk];
    dst.u[v] = (unsigned)yp[0] | ((unsigned)yp[1] << 16);
  }
}

// A fragment from global u (causal window ending at t), f32 -> f16.
// Branchless edge handling: clamp index to a valid address, mask the value.
__device__ __forceinline__ void load_a_u(const float* __restrict__ u,
                                         int t, int brow, int hi, int kt,
                                         FragU& dst) {
#pragma unroll
  for (int v = 0; v < 8; ++v) {
    int klocal = (hi ? 8 : 0) + ((v >= 4) ? 16 : 0) + 2 * (v & 3);
    int kg = kt * 32 + klocal;
    int c  = kg >> 6;
    int kk = kg & 63;
    int tt  = t - (RF - 1) + kk;
    int tt0 = tt     < 0 ? 0 : tt;
    int tt1 = tt + 1 < 0 ? 0 : tt + 1;
    const float* up = &u[(brow * NIN + c) * T_SZ];
    float x0 = up[tt0];
    float x1 = up[tt1];
    x0 = (tt     >= 0) ? x0 : 0.0f;   // v_cndmask, no exec branching
    x1 = (tt + 1 >= 0) ? x1 : 0.0f;
    dst.u[v] = f2h_bits(x0) | (f2h_bits(x1) << 16);
  }
}

__device__ __forceinline__ void load_b(const unsigned* __restrict__ frag,
                                       int kt, int nt, int lane, FragU& dst) {
  const unsigned* bp = &frag[((kt * NT_TILES + nt) * 32 + lane) * 8];
  dst.q[0] = *(const uint4*)bp;
  dst.q[1] = *(const uint4*)(bp + 4);
}

// -------------------------------------------------------------------------
// Bake W1 (f32) into per-lane WMMA B-fragment layout (f16 pairs), for both
// the input half (Wu, channels 0..3) and recurrent half (Wy, channels 4..7).
// B fragment (16-bit, 32x16): lanes 0-15: N=lane, K 0-7 (v0..3), 16-23 (v4..7);
// lanes 16-31: N=lane-16, K 8-15 (v0..3), 24-31 (v4..7).  K_flat = c*RF + kk.
// Array layout: frag[kt][nt][lane][v] (dwords).
// -------------------------------------------------------------------------
__global__ void prep_frags(const float* __restrict__ W1,
                           unsigned* __restrict__ wu_frag,
                           unsigned* __restrict__ wy_frag) {
  int idx = blockIdx.x * blockDim.x + threadIdx.x;
  if (idx >= FRAG_DWORDS) return;
  int v    = idx & 7;
  int lane = (idx >> 3) & 31;
  int nt   = (idx >> 8) & 15;
  int kt   = idx >> 12;

  int n      = nt * 16 + (lane & 15);
  int klocal = ((lane >= 16) ? 8 : 0) + ((v >= 4) ? 16 : 0) + 2 * (v & 3);
  int kg     = kt * 32 + klocal;      // even, so kg and kg+1 share a channel
  int c      = kg >> 6;
  int kk     = kg & 63;

  float u0 = W1[(n * NCH + c) * RF + kk];
  float u1 = W1[(n * NCH + c) * RF + kk + 1];
  float y0 = W1[(n * NCH + NIN + c) * RF + kk];
  float y1 = W1[(n * NCH + NIN + c) * RF + kk + 1];

  wu_frag[idx] = f2h_bits(u0) | (f2h_bits(u1) << 16);
  wy_frag[idx] = f2h_bits(y0) | (f2h_bits(y1) << 16);
}

// -------------------------------------------------------------------------
// Parallel precompute of the input-conv half for all t:
//   h_pre (fragment-major: [t][wave][lane][64]) = b1 + Wu * u_win(t)
// One block per t; Wu fragments staged to LDS with async copies.
// -------------------------------------------------------------------------
__global__ __launch_bounds__(512) void conv_kernel(const float* __restrict__ u,
                                                   const float* __restrict__ b1,
                                                   const unsigned* __restrict__ wu_frag,
                                                   float* __restrict__ h_pre) {
  extern __shared__ char csm[];
  unsigned* wuf = (unsigned*)csm;               // 32768 dwords = 128KB

  const int tid  = threadIdx.x;
  const int t    = blockIdx.x;
  const int lane = tid & 31;
  const int w    = tid >> 5;           // 0..15
  const int mt   = w & 7;
  const int ng   = w >> 3;             // 0..1 (8 n-tiles each)
  const int col  = lane & 15;
  const int hi   = lane >> 4;
  const int brow = mt * 16 + col;      // A-matrix row (M = lane&15)

  // Stage Wu fragments into LDS (async, ASYNCcnt-tracked).
  {
    unsigned lds_base = (unsigned)(uintptr_t)(void*)wuf;
    for (int i = tid; i < FRAG_DWORDS / 4; i += 512)
      async_g2l_b128(lds_base + (unsigned)i * 16u, (unsigned)i * 16u, wu_frag);
    wait_async0();
  }

  AccU acc[8];
#pragma unroll
  for (int j = 0; j < 8; ++j) {
    float bv = b1[(ng * 8 + j) * 16 + col];
#pragma unroll
    for (int r = 0; r < 8; ++r) acc[j].f[r] = bv;
  }

  // A fragments ping-pong by kt parity (16 regs instead of 64).
  FragU ab[2];
  load_a_u(u, t, brow, hi, 0, ab[0]);

  __syncthreads();   // LDS staging visible to all waves

  // Per-kt batched B fragments: all 8 live at once (64 distinct regs), each
  // reloaded for kt+1 right after its WMMA consumes it -> partial DS waits.
  FragU bbuf[8];
#pragma unroll
  for (int j = 0; j < 8; ++j) load_b(wuf, 0, ng * 8 + j, lane, bbuf[j]);

#pragma unroll
  for (int kt = 0; kt < KT_TILES; ++kt) {
    if (kt + 1 < KT_TILES) load_a_u(u, t, brow, hi, kt + 1, ab[(kt + 1) & 1]);
#pragma unroll
    for (int j = 0; j < 8; ++j) {
      acc[j].v = __builtin_amdgcn_wmma_f32_16x16x32_f16(
          false, ab[kt & 1].v, false, bbuf[j].v, (short)0, acc[j].v, false, false);
      if (kt + 1 < KT_TILES) load_b(wuf, kt + 1, ng * 8 + j, lane, bbuf[j]);
    }
  }

  // Fragment-major store: this lane's 64 floats are contiguous.
  float4* hps = (float4*)(h_pre + (((size_t)t * 16 + w) * 32 + lane) * 64);
#pragma unroll
  for (int j = 0; j < 8; ++j) {
    hps[2 * j]     = *(const float4*)&acc[j].f[0];
    hps[2 * j + 1] = *(const float4*)&acc[j].f[4];
  }
}

// -------------------------------------------------------------------------
// Sequential recurrence: single persistent workgroup (1 WGP), 16 waves.
// LDS: ybuf[4][128][128] f16 double-written ring (128KB) + Wy fragments
// (128KB, async-staged) + yacc[128][4] f32 (2KB).  <= 320KB per workgroup.
// -------------------------------------------------------------------------
__global__ __launch_bounds__(512) void seq_kernel(const float* __restrict__ h_pre,
                                                  const unsigned* __restrict__ wy_frag_g,
                                                  const float* __restrict__ W2,
                                                  const float* __restrict__ b2,
                                                  float* __restrict__ out) {
  extern __shared__ char smem[];
  unsigned short* ybuf = (unsigned short*)smem;            // 4*128*128 shorts
  unsigned*       wyf  = (unsigned*)(smem + 131072);       // 32768 dwords
  float*          yacc = (float*)(smem + 262144);          // 512 floats

  const int tid  = threadIdx.x;
  const int lane = tid & 31;
  const int w    = tid >> 5;
  const int mt   = w & 7;
  const int ng   = w >> 3;
  const int col  = lane & 15;
  const int hi   = lane >> 4;
  const int brow = mt * 16 + col;

  // Async-stage Wy fragments into LDS; zero ring + reduction buffers.
  {
    unsigned lds_base = (unsigned)(uintptr_t)(void*)wyf;
    for (int i = tid; i < FRAG_DWORDS / 4; i += 512)
      async_g2l_b128(lds_base + (unsigned)i * 16u, (unsigned)i * 16u, wy_frag_g);
  }
  for (int i = tid; i < 4 * B_SZ * 128 / 2; i += 512) ((unsigned*)ybuf)[i] = 0u;
  yacc[tid] = 0.0f;
  wait_async0();

  // Per-lane W2 columns: w2l[j][o] = W2[o][ (ng*8+j)*16 + col ]
  float w2l[8][4];
#pragma unroll
  for (int j = 0; j < 8; ++j)
#pragma unroll
    for (int o = 0; o < 4; ++o)
      w2l[j][o] = W2[o * HID + (ng * 8 + j) * 16 + col];

  const int   bw  = tid >> 2;     // writer mapping: batch row
  const int   ow  = tid & 3;      // writer mapping: output channel
  const float b2v = b2[ow];

  __syncthreads();

  for (int t = 0; t < T_SZ; ++t) {
    const int base = t & (RF - 1);

    // Prefetch next step's per-lane h_pre region (256B) into caches.
    if (t + 1 < T_SZ) {
      const float* pf = h_pre + (((size_t)(t + 1) * 16 + w) * 32 + lane) * 64;
      __builtin_prefetch(pf, 0, 0);
      __builtin_prefetch(pf + 32, 0, 0);
    }

    // A fragments ping-pong by kt parity; B fragments batched per kt.
    FragU ab[2];
    load_a_ring(ybuf, brow, base, hi, 0, ab[0]);
    FragU bbuf[8];
#pragma unroll
    for (int j = 0; j < 8; ++j) load_b(wyf, 0, ng * 8 + j, lane, bbuf[j]);

    AccU acc[8];
#pragma unroll
    for (int kt = 0; kt < KT_TILES; ++kt) {
      if (kt + 1 < KT_TILES)
        load_a_ring(ybuf, brow, base, hi, kt + 1, ab[(kt + 1) & 1]);
#pragma unroll
      for (int j = 0; j < 8; ++j) {
        if (kt == 0) {
          v8f z = {};   // C operand: fresh zero (foldable to inline 0)
          acc[j].v = __builtin_amdgcn_wmma_f32_16x16x32_f16(
              false, ab[0].v, false, bbuf[j].v, (short)0, z, false, false);
        } else {
          acc[j].v = __builtin_amdgcn_wmma_f32_16x16x32_f16(
              false, ab[kt & 1].v, false, bbuf[j].v, (short)0, acc[j].v, false, false);
        }
        if (kt + 1 < KT_TILES) load_b(wyf, kt + 1, ng * 8 + j, lane, bbuf[j]);
      }
    }

    // Fold in precomputed input-conv + bias (16 x b128 loads, contiguous).
    {
      const float4* hp = (const float4*)(h_pre + (((size_t)t * 16 + w) * 32 + lane) * 64);
#pragma unroll
      for (int j = 0; j < 8; ++j) {
        float4 h0 = hp[2 * j], h1 = hp[2 * j + 1];
        acc[j].f[0] += h0.x; acc[j].f[1] += h0.y;
        acc[j].f[2] += h0.z; acc[j].f[3] += h0.w;
        acc[j].f[4] += h1.x; acc[j].f[5] += h1.y;
        acc[j].f[6] += h1.z; acc[j].f[7] += h1.w;
      }
    }

    // Epilogue: y_partial[b][o] = sum_n relu(h[b][n]) * W2[o][n]
#pragma unroll
    for (int o = 0; o < 4; ++o) {
#pragma unroll
      for (int r = 0; r < 8; ++r) {
        float p = 0.0f;
#pragma unroll
        for (int j = 0; j < 8; ++j) {
          float hv = acc[j].f[r];
          hv = hv > 0.0f ? hv : 0.0f;
          p += hv * w2l[j][o];
        }
        // Reduce over the 16 lanes holding one row (xor masks stay in-half).
        p += __shfl_xor(p, 1, 32);
        p += __shfl_xor(p, 2, 32);
        p += __shfl_xor(p, 4, 32);
        p += __shfl_xor(p, 8, 32);
        if (col == 0) {
          int row = mt * 16 + hi * 8 + r;
          atomicAdd(&yacc[row * NOUT + o], p);   // ds_add_f32
        }
      }
    }

    __syncthreads();

    // One thread per (b, o): finalize y_t, emit output, feed the ring.
    {
      float y = yacc[tid] + b2v;
      out[((size_t)bw * NOUT + ow) * T_SZ + t] = y;
      unsigned short yh = (unsigned short)f2h_bits(y);
      int sbase = (ow * B_SZ + bw) * 128;
      ybuf[sbase + base]      = yh;   // primary slot (t mod 64)
      ybuf[sbase + base + 64] = yh;   // duplicate for contiguous windows
      yacc[tid] = 0.0f;               // re-zero for next step
    }

    __syncthreads();
  }
}

extern "C" void kernel_launch(void* const* d_in, const int* in_sizes, int n_in,
                              void* d_out, int out_size, void* d_ws, size_t ws_size,
                              hipStream_t stream) {
  (void)in_sizes; (void)n_in; (void)out_size;
  const float* u  = (const float*)d_in[0];
  const float* W1 = (const float*)d_in[1];
  const float* b1 = (const float*)d_in[2];
  const float* W2 = (const float*)d_in[3];
  const float* b2 = (const float*)d_in[4];
  float* out = (float*)d_out;

  const size_t frag_bytes = (size_t)FRAG_DWORDS * 4;     // 128KB
  const size_t off_wu = 0;
  const size_t off_wy = frag_bytes;
  const size_t off_h  = 2 * frag_bytes;
  const size_t need   = off_h + (size_t)T_SZ * B_SZ * HID * sizeof(float);
  if (ws_size < need) return;  // workspace too small for precompute path

  unsigned* wu_frag = (unsigned*)((char*)d_ws + off_wu);
  unsigned* wy_frag = (unsigned*)((char*)d_ws + off_wy);
  float*    h_pre   = (float*)((char*)d_ws + off_h);

  // 1) Bake weight fragments (f32 -> f16, WMMA B-layout).
  prep_frags<<<(FRAG_DWORDS + 255) / 256, 256, 0, stream>>>(W1, wu_frag, wy_frag);

  // 2) Parallel input-convolution precompute across all T (throughput phase).
  conv_kernel<<<T_SZ, 512, 131072, stream>>>(u, b1, wu_frag, h_pre);

  // 3) Latency-bound recurrence on one persistent workgroup (one WGP).
  const size_t smem = 131072 /*ybuf*/ + 131072 /*wyf*/ + 2048 /*yacc*/;
  seq_kernel<<<1, 512, smem, stream>>>(h_pre, wy_frag, W2, b2, out);
}
